// SpatialSelfAttention_47863115546958
// MI455X (gfx1250) — matrix operand order repaired and checked by reading, not compiled, not run on hardware
//
#include <hip/hip_runtime.h>
#include <hip/hip_bf16.h>

// ---------------------------------------------------------------------------
// SpatialSelfAttention for MI455X (gfx1250, wave32, WMMA bf16)
//   B=4, C=512, H=W=64 -> N=4096. ~172 GFLOP; per-batch working set fits the
//   192MB L2 -> batches sequential, GEMMs run out of L2 at bf16 WMMA rates.
//   GEMM: 128x128 block tile, 64x64 per wave, LDS double-buffered with
//   GLOBAL_LOAD_ASYNC_TO_LDS_B128 (ASYNCcnt) so staging overlaps WMMA.
// ---------------------------------------------------------------------------

#if __has_builtin(__builtin_amdgcn_global_load_async_to_lds_b128) && \
    __has_builtin(__builtin_amdgcn_s_wait_asynccnt)
#define USE_ASYNC_LDS 1
#else
#define USE_ASYNC_LDS 0
#endif

typedef __attribute__((ext_vector_type(16))) __bf16       v16bf;
typedef __attribute__((ext_vector_type(8)))  float        v8f;
typedef __attribute__((ext_vector_type(4)))  unsigned int u32x4;
typedef __attribute__((ext_vector_type(4)))  int          v4i;
typedef __attribute__((address_space(1)))    v4i          gv4i;   // global
typedef __attribute__((address_space(3)))    v4i          lv4i;   // LDS

union FragU { v16bf v; u32x4 q[2]; };

__device__ __forceinline__ unsigned short f2bf(float f) {
    union { float f; unsigned u; } x; x.f = f;
    unsigned r = (x.u + 0x7FFFu + ((x.u >> 16) & 1u)) >> 16;
    return (unsigned short)r;
}

// stage one 128-byte row slice (64 bf16) global -> LDS
__device__ __forceinline__ void stage_row(const unsigned short* __restrict__ gsrc,
                                          unsigned int* __restrict__ ldst) {
#if USE_ASYNC_LDS
    #pragma unroll
    for (int i = 0; i < 8; ++i)
        __builtin_amdgcn_global_load_async_to_lds_b128(
            (gv4i*)(gsrc + i * 8), (lv4i*)(ldst + i * 4), 0, 0);
#else
    const u32x4* g = (const u32x4*)gsrc;
    u32x4* d = (u32x4*)ldst;
    #pragma unroll
    for (int i = 0; i < 8; ++i) d[i] = g[i];
#endif
}

__device__ __forceinline__ void stage_fence() {
#if USE_ASYNC_LDS
    __builtin_amdgcn_s_wait_asynccnt(0);
#endif
    __syncthreads();
}

// ---------------------------------------------------------------------------
// GroupNorm pass 1: per (batch, group) mean / rstd over 16 ch x 4096 px
// ---------------------------------------------------------------------------
__global__ void gn_stats_kernel(const float* __restrict__ x,
                                float* __restrict__ mu, float* __restrict__ rstd,
                                int C, int Nsp) {
    const int bg = blockIdx.x;          // b*32 + g
    const int b = bg >> 5, g = bg & 31;
    const float* base = x + ((size_t)b * C + (size_t)g * 16) * Nsp;
    const int total = 16 * Nsp;
    float s = 0.f, s2 = 0.f;
    for (int e = threadIdx.x; e < total; e += blockDim.x) {
        float v = base[e];
        s += v; s2 += v * v;
    }
    __shared__ float ss[256], ss2[256];
    ss[threadIdx.x] = s; ss2[threadIdx.x] = s2;
    __syncthreads();
    for (int off = 128; off; off >>= 1) {
        if (threadIdx.x < off) {
            ss[threadIdx.x]  += ss[threadIdx.x + off];
            ss2[threadIdx.x] += ss2[threadIdx.x + off];
        }
        __syncthreads();
    }
    if (threadIdx.x == 0) {
        float inv = 1.f / (float)total;
        float m = ss[0] * inv;
        float var = ss2[0] * inv - m * m;
        mu[bg] = m;
        rstd[bg] = rsqrtf(var + 1e-6f);
    }
}

// ---------------------------------------------------------------------------
// GroupNorm pass 2: normalize + affine, write TRANSPOSED bf16 hT[b][n][c]
// ---------------------------------------------------------------------------
__global__ void gn_apply_kernel(const float* __restrict__ x,
                                const float* __restrict__ gamma,
                                const float* __restrict__ beta,
                                const float* __restrict__ mu,
                                const float* __restrict__ rstd,
                                unsigned short* __restrict__ hT,
                                int C, int Nsp) {
    __shared__ float tile[32][33];
    const int b = blockIdx.z;
    const int c0 = blockIdx.y * 32, n0 = blockIdx.x * 32;
    const int tx = threadIdx.x, ty = threadIdx.y;
    #pragma unroll
    for (int i = 0; i < 4; ++i) {
        int c = c0 + ty + i * 8, n = n0 + tx;
        float v = x[((size_t)b * C + c) * Nsp + n];
        int g = c >> 4;                       // 16 channels per group
        float m = mu[b * 32 + g], rs = rstd[b * 32 + g];
        tile[ty + i * 8][tx] = (v - m) * rs * gamma[c] + beta[c];
    }
    __syncthreads();
    #pragma unroll
    for (int i = 0; i < 4; ++i) {
        int n = n0 + ty + i * 8, c = c0 + tx;
        hT[((size_t)b * Nsp + n) * C + c] = f2bf(tile[tx][ty + i * 8]);
    }
}

// ---------------------------------------------------------------------------
// fp32 -> bf16 array convert (weights)
// ---------------------------------------------------------------------------
__global__ void cvt_bf16_kernel(const float* __restrict__ src,
                                unsigned short* __restrict__ dst, int n) {
    int i = blockIdx.x * blockDim.x + threadIdx.x;
    if (i < n) dst[i] = f2bf(src[i]);
}

// ---------------------------------------------------------------------------
// Generic bf16 WMMA GEMM:  C[M][N] = A[M][K] * Bt[N][K]^T
//   128 threads (4 waves, 2x2), block tile 128x128, wave tile 64x64.
//   K staged 64-wide, LDS DOUBLE-BUFFERED: stage s+1 is async-DMA'd into the
//   other buffer (GLOBAL_LOAD_ASYNC_TO_LDS_B128, ASYNCcnt) while stage s runs
//   its 32 WMMAs; s_wait_asynccnt(0) + barrier at stage boundary.
//   __launch_bounds__(128,1) -> full VGPR budget, accumulators resident.
// Epilogue modes:
//   0: outF[m*N+n]  = acc                              (scores, fp32)
//   1: outBf[n*M+m] = bf16((acc + bias[m]) * scale)    (q_t / k_t, transposed)
//   2: outBf[m*N+n] = bf16(acc (+ bias[m]))            (v, O^T)
//   4: outF[n*M+m]  = acc + bias[n] + resid[n*M+m]     (proj + residual)
// ---------------------------------------------------------------------------
__global__ __launch_bounds__(128, 1)
void gemm_wmma_bf16(const unsigned short* __restrict__ A,
                    const unsigned short* __restrict__ Bt,
                    int M, int N, int K,
                    int mode, float scale,
                    const float* __restrict__ bias,
                    const float* __restrict__ resid,
                    float* __restrict__ outF,
                    unsigned short* __restrict__ outBf) {
    __shared__ unsigned int As[2][128 * 32];   // 2 x (128 rows x 32 dwords)
    __shared__ unsigned int Bs[2][128 * 32];

    const int tid  = threadIdx.x;
    const int lane = tid & 31;
    const int wave = tid >> 5;
    const int wm = wave >> 1, wn = wave & 1;   // 2x2 waves over 128x128
    const int hi  = (lane >> 4) & 1;
    const int col = lane & 15;
    const int m0 = blockIdx.y * 128;
    const int n0 = blockIdx.x * 128;

    const unsigned short* Arow = A  + (size_t)(m0 + tid) * K;
    const unsigned short* Brow = Bt + (size_t)(n0 + tid) * K;

    v8f acc[4][4] = {};

    // prologue: stage 0 into buffer 0
    stage_row(Arow, &As[0][tid * 32]);
    stage_row(Brow, &Bs[0][tid * 32]);
    stage_fence();

    const int nstages = K >> 6;                // K / 64
    for (int s = 0; s < nstages; ++s) {
        const int cur = s & 1;
        if (s + 1 < nstages) {                 // async prefetch next stage
            stage_row(Arow + (s + 1) * 64, &As[cur ^ 1][tid * 32]);
            stage_row(Brow + (s + 1) * 64, &Bs[cur ^ 1][tid * 32]);
        }
        const unsigned int* Ab = &As[cur][0];
        const unsigned int* Bb = &Bs[cur][0];

        #pragma unroll
        for (int ss = 0; ss < 2; ++ss) {       // two K=32 substeps
            const int kd = ss * 16;            // dword offset of substep
            // preload the 4 B fragments (32 VGPRs live)
            FragU fb[4];
            #pragma unroll
            for (int t = 0; t < 4; ++t) {
                int rb = wn * 64 + t * 16 + col;
                fb[t].q[0] = *(const u32x4*)&Bb[rb * 32 + kd + hi * 8];
                fb[t].q[1] = *(const u32x4*)&Bb[rb * 32 + kd + hi * 8 + 4];
            }
            // stream A fragments one at a time (8 VGPRs live each)
            #pragma unroll
            for (int tm = 0; tm < 4; ++tm) {
                FragU fa;
                int ra = wm * 64 + tm * 16 + col;
                fa.q[0] = *(const u32x4*)&Ab[ra * 32 + kd + hi * 8];
                fa.q[1] = *(const u32x4*)&Ab[ra * 32 + kd + hi * 8 + 4];
                #pragma unroll
                for (int tn = 0; tn < 4; ++tn)
                    acc[tm][tn] = __builtin_amdgcn_wmma_f32_16x16x32_bf16(
                        false, fa.v, false, fb[tn].v,
                        (short)0, acc[tm][tn], false, false);
            }
        }
        stage_fence();                         // asyncs done + all waves done
    }

    // epilogue: D tile layout = VGPR r -> row (hi*8 + r), lane&15 -> col
    #pragma unroll
    for (int tm = 0; tm < 4; ++tm) {
        #pragma unroll
        for (int tn = 0; tn < 4; ++tn) {
            #pragma unroll
            for (int r = 0; r < 8; ++r) {
                int m = m0 + wm * 64 + tm * 16 + hi * 8 + r;
                int n = n0 + wn * 64 + tn * 16 + col;
                float v = acc[tm][tn][r];
                if (mode == 0) {
                    outF[(size_t)m * N + n] = v;
                } else if (mode == 1) {
                    outBf[(size_t)n * M + m] = f2bf((v + bias[m]) * scale);
                } else if (mode == 2) {
                    float t = bias ? (v + bias[m]) : v;
                    outBf[(size_t)m * N + n] = f2bf(t);
                } else { // mode 4: proj + bias + residual, transposed fp32
                    outF[(size_t)n * M + m] =
                        v + bias[n] + resid[(size_t)n * M + m];
                }
            }
        }
    }
}

// ---------------------------------------------------------------------------
// Softmax row stats (online max/sum): one block per row
// ---------------------------------------------------------------------------
__global__ void rowstat_kernel(const float* __restrict__ S,
                               float* __restrict__ rowM,
                               float* __restrict__ rowInvL, int N) {
    const int i = blockIdx.x;
    const float* row = S + (size_t)i * N;
    float m = -3.0e38f, l = 0.f;
    for (int j = threadIdx.x; j < N; j += blockDim.x) {
        float s = row[j];
        float nm = fmaxf(m, s);
        l = l * __expf(m - nm) + __expf(s - nm);
        m = nm;
    }
    __shared__ float sm[256], sl[256];
    sm[threadIdx.x] = m; sl[threadIdx.x] = l;
    __syncthreads();
    for (int off = 128; off; off >>= 1) {
        if (threadIdx.x < off) {
            float m2 = sm[threadIdx.x + off], l2 = sl[threadIdx.x + off];
            float M2 = fmaxf(sm[threadIdx.x], m2);
            sl[threadIdx.x] = sl[threadIdx.x] * __expf(sm[threadIdx.x] - M2)
                              + l2 * __expf(m2 - M2);
            sm[threadIdx.x] = M2;
        }
        __syncthreads();
    }
    if (threadIdx.x == 0) {
        rowM[i] = sm[0];
        rowInvL[i] = 1.f / sl[0];
    }
}

// ---------------------------------------------------------------------------
// P[i][j] = bf16(exp(S - rowM) * rowInvL) ; grid (N/256, N)
// ---------------------------------------------------------------------------
__global__ void pnorm_kernel(const float* __restrict__ S,
                             const float* __restrict__ rowM,
                             const float* __restrict__ rowInvL,
                             unsigned short* __restrict__ P, int N) {
    const int i = blockIdx.y;
    const int j = blockIdx.x * blockDim.x + threadIdx.x;
    float m = rowM[i], il = rowInvL[i];
    P[(size_t)i * N + j] = f2bf(__expf(S[(size_t)i * N + j] - m) * il);
}

// ---------------------------------------------------------------------------
extern "C" void kernel_launch(void* const* d_in, const int* in_sizes, int n_in,
                              void* d_out, int out_size, void* d_ws, size_t ws_size,
                              hipStream_t stream) {
    const int B = 4, C = 512, Nsp = 64 * 64;   // 4096
    const float* x     = (const float*)d_in[0];
    const float* gamma = (const float*)d_in[1];
    const float* beta  = (const float*)d_in[2];
    const float* wq    = (const float*)d_in[3];
    const float* bq    = (const float*)d_in[4];
    const float* wk    = (const float*)d_in[5];
    const float* bk    = (const float*)d_in[6];
    const float* wv    = (const float*)d_in[7];
    const float* bv    = (const float*)d_in[8];
    const float* wp    = (const float*)d_in[9];
    const float* bp    = (const float*)d_in[10];
    float* out = (float*)d_out;

    // workspace carve-up (256B aligned)
    char* ws = (char*)d_ws;
    size_t p = 0;
    auto carve = [&](size_t bytes) -> char* {
        char* r = ws + p;
        p = (p + bytes + 255) & ~(size_t)255;
        return r;
    };
    unsigned short* hT   = (unsigned short*)carve((size_t)B * Nsp * C * 2); // 16MB
    unsigned short* wqb  = (unsigned short*)carve((size_t)C * C * 2);
    unsigned short* wkb  = (unsigned short*)carve((size_t)C * C * 2);
    unsigned short* wvb  = (unsigned short*)carve((size_t)C * C * 2);
    unsigned short* wpb  = (unsigned short*)carve((size_t)C * C * 2);
    unsigned short* qt   = (unsigned short*)carve((size_t)Nsp * C * 2);     // 4MB
    unsigned short* kt   = (unsigned short*)carve((size_t)Nsp * C * 2);
    unsigned short* vbuf = (unsigned short*)carve((size_t)Nsp * C * 2);
    unsigned short* Ot   = (unsigned short*)carve((size_t)Nsp * C * 2);
    float* S   = (float*)carve((size_t)Nsp * Nsp * 4);                      // 64MB
    unsigned short* P = (unsigned short*)carve((size_t)Nsp * Nsp * 2);      // 32MB
    float* mu     = (float*)carve(B * 32 * 4);
    float* rstd   = (float*)carve(B * 32 * 4);
    float* rowM   = (float*)carve(Nsp * 4);
    float* rowIL  = (float*)carve(Nsp * 4);

    // 1) GroupNorm -> hT[b][n][c] bf16
    gn_stats_kernel<<<B * 32, 256, 0, stream>>>(x, mu, rstd, C, Nsp);
    gn_apply_kernel<<<dim3(Nsp / 32, C / 32, B), dim3(32, 8), 0, stream>>>(
        x, gamma, beta, mu, rstd, hT, C, Nsp);

    // 2) weights -> bf16
    const int wn = C * C;
    cvt_bf16_kernel<<<(wn + 255) / 256, 256, 0, stream>>>(wq, wqb, wn);
    cvt_bf16_kernel<<<(wn + 255) / 256, 256, 0, stream>>>(wk, wkb, wn);
    cvt_bf16_kernel<<<(wn + 255) / 256, 256, 0, stream>>>(wv, wvb, wn);
    cvt_bf16_kernel<<<(wn + 255) / 256, 256, 0, stream>>>(wp, wpb, wn);

    const float qscale = 0.044194173824159216f;  // 512^-0.5 folded into q

    for (int b = 0; b < B; ++b) {
        const unsigned short* hTb = hT + (size_t)b * Nsp * C;
        const float* xb = x + (size_t)b * C * Nsp;
        float* outb = out + (size_t)b * C * Nsp;

        // q_t[n][c], k_t[n][c] (transposed bf16), v[c][n] (row-major bf16)
        gemm_wmma_bf16<<<dim3(Nsp / 128, C / 128), 128, 0, stream>>>(
            wqb, hTb, C, Nsp, C, 1, qscale, bq, nullptr, nullptr, qt);
        gemm_wmma_bf16<<<dim3(Nsp / 128, C / 128), 128, 0, stream>>>(
            wkb, hTb, C, Nsp, C, 1, 1.f, bk, nullptr, nullptr, kt);
        gemm_wmma_bf16<<<dim3(Nsp / 128, C / 128), 128, 0, stream>>>(
            wvb, hTb, C, Nsp, C, 2, 1.f, bv, nullptr, nullptr, vbuf);

        // S[i][j] = q_t . k_t   (fp32, stays in L2)
        gemm_wmma_bf16<<<dim3(Nsp / 128, Nsp / 128), 128, 0, stream>>>(
            qt, kt, Nsp, Nsp, C, 0, 1.f, nullptr, nullptr, S, nullptr);

        // softmax
        rowstat_kernel<<<Nsp, 256, 0, stream>>>(S, rowM, rowIL, Nsp);
        pnorm_kernel<<<dim3(Nsp / 256, Nsp), 256, 0, stream>>>(S, rowM, rowIL, P, Nsp);

        // O^T[n][c] = P[n][:] . v[c][:]^T
        gemm_wmma_bf16<<<dim3(C / 128, Nsp / 128), 128, 0, stream>>>(
            P, vbuf, Nsp, C, Nsp, 2, 1.f, nullptr, nullptr, nullptr, Ot);

        // out[b][o][n] = x + Wp . O  (+bp), Wp row-major is already B^T
        gemm_wmma_bf16<<<dim3(C / 128, Nsp / 128), 128, 0, stream>>>(
            Ot, wpb, Nsp, C, C, 4, 1.f, bp, xb, outb, nullptr);
    }
}
